// AudioTemporalConsistencyModule_46437186404923
// MI455X (gfx1250) — matrix operand order repaired and checked by reading, not compiled, not run on hardware
//
#include <hip/hip_runtime.h>
#include <hip/hip_bf16.h>

typedef __attribute__((ext_vector_type(16))) unsigned short v16u;
typedef __attribute__((ext_vector_type(16))) __bf16         v16bf;
typedef __attribute__((ext_vector_type(8)))  float          v8f;

static __device__ __forceinline__ unsigned short f2bf(float f) {
    unsigned int u = __float_as_uint(f);
    u += 0x7FFFu + ((u >> 16) & 1u);          // round-to-nearest-even
    return (unsigned short)(u >> 16);
}
static __device__ __forceinline__ float bf2f(unsigned short h) {
    return __uint_as_float(((unsigned int)h) << 16);
}

static __device__ __forceinline__ v8f wmma_bf16(v16u a, v16u b, v8f c) {
    return __builtin_amdgcn_wmma_f32_16x16x32_bf16(
        false, __builtin_bit_cast(v16bf, a),
        false, __builtin_bit_cast(v16bf, b),
        (short)0, c, false, false);
}

// ---------------------------------------------------------------------------
// Repack W[f32, K x N row-major] into hi/lo bf16 planes laid out as this code's
// per-lane B fragment for v_wmma_*_16x16x32_bf16 (contiguous k order, not AMD's table order):
//   block(tile t of 16 cols, kb of 32 K) is 512 ushorts; lane l holds 16
//   contiguous values: n = t*16 + (l&15), k = kb*32 + 16*(l>>4) + j, j=0..15.
// ---------------------------------------------------------------------------
__global__ __launch_bounds__(256) void pack_w_kernel(
    const float* __restrict__ W, unsigned short* __restrict__ Hi,
    unsigned short* __restrict__ Lo, int K, int N)
{
    int idx = blockIdx.x * blockDim.x + threadIdx.x;
    if (idx >= K * N) return;
    int j   = idx & 15;
    int l   = (idx >> 4) & 31;
    int blk = idx >> 9;
    int KB  = K >> 5;
    int kb  = blk % KB;
    int t   = blk / KB;
    int n   = t * 16 + (l & 15);
    int k   = kb * 32 + ((l >> 4) << 4) + j;
    float f = W[(size_t)k * N + n];
    unsigned short h = f2bf(f);
    Hi[idx] = h;
    Lo[idx] = f2bf(f - bf2f(h));
}

// ---------------------------------------------------------------------------
// Fused GEMM (bf16 hi/lo compensated, f32 accumulate) + bias + LayerNorm +
// exact GELU.  One workgroup = 16 rows x NCOLS.  8 waves; wave owns NT tiles
// of 16x16.  KBLK = K/32 K-blocks.
// PAIRS: A row m reads x at (m + m/511)*1024 (implicit concat of frames).
// ---------------------------------------------------------------------------
template <int NT, int KBLK, int NCOLS, bool PAIRS>
__global__ __launch_bounds__(256) void fused_gemm_ln_gelu(
    const float* __restrict__ A,
    const unsigned short* __restrict__ Bhi,
    const unsigned short* __restrict__ Blo,
    const float* __restrict__ bias,
    const float* __restrict__ gamma,
    const float* __restrict__ beta,
    float* __restrict__ Out)
{
    const int lane = threadIdx.x & 31;
    const int wave = threadIdx.x >> 5;
    const int hi   = lane >> 4;
    const int nl   = lane & 15;

    // A row held by this lane (A fragment: M = lane&15)
    int row_g = blockIdx.x * 16 + nl;
    size_t aoff;
    if (PAIRS) {
        int b = row_g / 511;                       // batch index
        aoff  = (size_t)(row_g + b) * 1024;        // x[b, s : s+2, :] contiguous
    } else {
        aoff  = (size_t)row_g * (size_t)(KBLK * 32);
    }
    const float* arow = A + aoff;

    v8f acc[NT];
#pragma unroll
    for (int i = 0; i < NT; ++i) acc[i] = v8f{};

    for (int kb = 0; kb < KBLK; ++kb) {
        // A fragment: lane half `hi` holds K = 32kb + 16hi + [0..15]
        // (matches packed B layout), contiguous in memory -> float4 loads.
        int k0 = kb * 32 + 16 * hi;
        float4 q0 = *(const float4*)(arow + k0);
        float4 q1 = *(const float4*)(arow + k0 + 4);
        float4 q2 = *(const float4*)(arow + k0 + 8);
        float4 q3 = *(const float4*)(arow + k0 + 12);
        if (kb + 1 < KBLK) __builtin_prefetch(arow + k0 + 32, 0, 0);

        float af[16] = {q0.x, q0.y, q0.z, q0.w, q1.x, q1.y, q1.z, q1.w,
                        q2.x, q2.y, q2.z, q2.w, q3.x, q3.y, q3.z, q3.w};
        v16u ahi, alo;
#pragma unroll
        for (int j = 0; j < 16; ++j) {
            unsigned short h = f2bf(af[j]);
            ahi[j] = h;
            alo[j] = f2bf(af[j] - bf2f(h));
        }

#pragma unroll
        for (int i = 0; i < NT; ++i) {
            size_t boff = ((size_t)(wave * NT + i) * KBLK + kb) * 512 + lane * 16;
            v16u bhi = *(const v16u*)(Bhi + boff);
            v16u blo = *(const v16u*)(Blo + boff);
            acc[i] = wmma_bf16(ahi, bhi, acc[i]);   // hi*hi
            acc[i] = wmma_bf16(ahi, blo, acc[i]);   // hi*lo
            acc[i] = wmma_bf16(alo, bhi, acc[i]);   // lo*hi
        }
    }

    // ---- bias, then LayerNorm statistics --------------------------------
    // D layout: lane holds (m = r + 8*hi, n = tile*16 + nl), r = 0..7.
#pragma unroll
    for (int i = 0; i < NT; ++i) {
        int n = (wave * NT + i) * 16 + nl;
        float bn = bias[n];
#pragma unroll
        for (int r = 0; r < 8; ++r) acc[i][r] += bn;
    }

    float ps[8], pq[8];
#pragma unroll
    for (int r = 0; r < 8; ++r) { ps[r] = 0.f; pq[r] = 0.f; }
#pragma unroll
    for (int i = 0; i < NT; ++i)
#pragma unroll
        for (int r = 0; r < 8; ++r) {
            float v = acc[i][r];
            ps[r] += v; pq[r] += v * v;
        }
    // reduce over the 16-lane n dimension (xor masks 1,2,4,8 stay in-half)
#pragma unroll
    for (int m = 1; m < 16; m <<= 1)
#pragma unroll
        for (int r = 0; r < 8; ++r) {
            ps[r] += __shfl_xor(ps[r], m, 32);
            pq[r] += __shfl_xor(pq[r], m, 32);
        }

    __shared__ float redS[8][16], redQ[8][16], muS[16], rsS[16];
    if (nl == 0) {
#pragma unroll
        for (int r = 0; r < 8; ++r) {
            redS[wave][8 * hi + r] = ps[r];
            redQ[wave][8 * hi + r] = pq[r];
        }
    }
    __syncthreads();
    if (threadIdx.x < 16) {
        float s = 0.f, q = 0.f;
#pragma unroll
        for (int w = 0; w < 8; ++w) { s += redS[w][threadIdx.x]; q += redQ[w][threadIdx.x]; }
        float mu  = s / (float)NCOLS;
        float var = q / (float)NCOLS - mu * mu;
        muS[threadIdx.x] = mu;
        rsS[threadIdx.x] = rsqrtf(var + 1e-5f);
    }
    __syncthreads();

    // ---- normalize, exact GELU, store -----------------------------------
#pragma unroll
    for (int i = 0; i < NT; ++i) {
        int n = (wave * NT + i) * 16 + nl;
        float g = gamma[n], bb = beta[n];
#pragma unroll
        for (int r = 0; r < 8; ++r) {
            int m = 8 * hi + r;
            float v = (acc[i][r] - muS[m]) * rsS[m] * g + bb;
            v = 0.5f * v * (1.0f + erff(v * 0.70710678118654752f));
            Out[(size_t)(blockIdx.x * 16 + m) * NCOLS + n] = v;
        }
    }
}

// ---------------------------------------------------------------------------
// scores = sigmoid(h2 @ W3 + b3)  — one wave per row, K = 512.
// ---------------------------------------------------------------------------
__global__ __launch_bounds__(256) void score_kernel(
    const float* __restrict__ H2, const float* __restrict__ W3,
    const float* __restrict__ b3, float* __restrict__ out, int M)
{
    int lane = threadIdx.x & 31;
    int wave = threadIdx.x >> 5;
    int row  = blockIdx.x * 8 + wave;
    if (row >= M) return;
    const float* hr = H2 + (size_t)row * 512;
    float s = 0.f;
#pragma unroll 4
    for (int j = lane; j < 512; j += 32) s += hr[j] * W3[j];
#pragma unroll
    for (int m = 16; m >= 1; m >>= 1) s += __shfl_xor(s, m, 32);
    if (lane == 0) {
        float z = s + b3[0];
        out[row] = 1.0f / (1.0f + expf(-z));
    }
}

extern "C" void kernel_launch(void* const* d_in, const int* in_sizes, int n_in,
                              void* d_out, int out_size, void* d_ws, size_t ws_size,
                              hipStream_t stream) {
    (void)in_sizes; (void)n_in; (void)out_size; (void)ws_size;
    const float* x   = (const float*)d_in[0];
    const float* W1  = (const float*)d_in[1];
    const float* b1  = (const float*)d_in[2];
    const float* g1  = (const float*)d_in[3];
    const float* be1 = (const float*)d_in[4];
    const float* W2  = (const float*)d_in[5];
    const float* b2  = (const float*)d_in[6];
    const float* g2  = (const float*)d_in[7];
    const float* be2 = (const float*)d_in[8];
    const float* W3  = (const float*)d_in[9];
    const float* b3  = (const float*)d_in[10];
    float* out = (float*)d_out;

    char* ws = (char*)d_ws;
    size_t off = 0;
    unsigned short* w1hi = (unsigned short*)(ws + off); off += (size_t)2048 * 1024 * 2;
    unsigned short* w1lo = (unsigned short*)(ws + off); off += (size_t)2048 * 1024 * 2;
    unsigned short* w2hi = (unsigned short*)(ws + off); off += (size_t)1024 * 512 * 2;
    unsigned short* w2lo = (unsigned short*)(ws + off); off += (size_t)1024 * 512 * 2;
    float* h1 = (float*)(ws + off); off += (size_t)16352 * 1024 * 4;
    float* h2 = (float*)(ws + off); off += (size_t)16352 * 512 * 4;

    // 1) repack weights into WMMA B-fragment hi/lo bf16 planes
    pack_w_kernel<<<(2048 * 1024) / 256, 256, 0, stream>>>(W1, w1hi, w1lo, 2048, 1024);
    pack_w_kernel<<<(1024 * 512) / 256, 256, 0, stream>>>(W2, w2hi, w2lo, 1024, 512);

    // 2) GEMM1 (K=2048, N=1024) + bias + LN + GELU   (16352/16 = 1022 blocks)
    fused_gemm_ln_gelu<8, 64, 1024, true ><<<1022, 256, 0, stream>>>(x,  w1hi, w1lo, b1, g1, be1, h1);
    // 3) GEMM2 (K=1024, N=512) + bias + LN + GELU
    fused_gemm_ln_gelu<4, 32, 512,  false><<<1022, 256, 0, stream>>>(h1, w2hi, w2lo, b2, g2, be2, h2);
    // 4) scores
    score_kernel<<<(16352 + 7) / 8, 256, 0, stream>>>(h2, W3, b3, out, 16352);
}
